// GraphAttentionLayer_77661598646342
// MI455X (gfx1250) — matrix operand order repaired and checked by reading, not compiled, not run on hardware
//
#include <hip/hip_runtime.h>
#include <cstdint>

// ---- constants for this problem ----
#define N_ROWS   8192
#define IN_F     256
#define OUT_F    128
#define HSZ      (N_ROWS * OUT_F)      // 1048576 elements of h
#define NSPLIT   4                     // j-splits in attention pass

typedef float v2f  __attribute__((ext_vector_type(2)));
typedef float v8f  __attribute__((ext_vector_type(8)));
typedef __bf16 v16bf __attribute__((ext_vector_type(16)));
typedef int   v4i  __attribute__((ext_vector_type(4)));

typedef __attribute__((address_space(1))) v4i* glb_v4i_p;
typedef __attribute__((address_space(3))) v4i* lds_v4i_p;

union BF16x16 { uint32_t u[8]; v16bf v; };

#if defined(__HIP_DEVICE_COMPILE__) && __has_builtin(__builtin_amdgcn_global_load_async_to_lds_b128)
#define HAVE_ASYNC_LDS 1
#else
#define HAVE_ASYNC_LDS 0
#endif

__device__ __forceinline__ void wait_async0() {
#if __has_builtin(__builtin_amdgcn_s_wait_asynccnt)
    __builtin_amdgcn_s_wait_asynccnt(0);
#else
    asm volatile("s_wait_asynccnt 0" ::: "memory");
#endif
}

__device__ __forceinline__ uint32_t f2bf(float f) {
    // round-to-nearest-even float32 -> bf16 (upper 16 bits)
    uint32_t u = __float_as_uint(f);
    u += 0x7FFFu + ((u >> 16) & 1u);
    return u >> 16;
}

// ---------------------------------------------------------------------------
// Kernel 1: h = X @ W^T + b   (fp32 WMMA 16x16x4, one 16x16 tile per wave)
// ---------------------------------------------------------------------------
__global__ void k_gemm1(const float* __restrict__ X, const float* __restrict__ Ww,
                        const float* __restrict__ Wb, float* __restrict__ h) {
    const int wave = (blockIdx.x * blockDim.x + threadIdx.x) >> 5;
    const int lane = threadIdx.x & 31;
    const int it = wave >> 3;          // 512 row tiles
    const int ot = wave & 7;           // 8 col tiles
    const int m  = lane & 15;
    const int kh = lane >> 4;
    const int row = it * 16 + m;       // A-matrix row for this lane
    const int col = ot * 16 + m;       // B/D column (n) for this lane

    const float* xp = X  + (size_t)row * IN_F + 2 * kh;
    const float* wp = Ww + (size_t)col * IN_F + 2 * kh;

    v8f acc = {};
    for (int k = 0; k < IN_F; k += 4) {
        // A layout (f32 16x4): VGPR v, lane l -> K = v + 2*(l>>4)
        v2f a, b;
        a[0] = xp[k];  a[1] = xp[k + 1];
        b[0] = wp[k];  b[1] = wp[k + 1];
        acc = __builtin_amdgcn_wmma_f32_16x16x4_f32(false, a, false, b,
                                                    (short)0, acc, false, false);
    }
    const float bias = Wb[col];
    // D layout: VGPR r, lane l -> M = r + 8*(l>>4), N = l&15
    for (int r = 0; r < 8; ++r)
        h[(size_t)(it * 16 + r + 8 * kh) * OUT_F + col] = acc[r] + bias;
}

// ---------------------------------------------------------------------------
// Kernel 2: per-row dots  s_src[i] = h[i,:].a_w[:F],  s_dst[i] = h[i,:].a_w[F:]
// ---------------------------------------------------------------------------
__global__ void k_dots(const float* __restrict__ h, const float* __restrict__ aw,
                       float* __restrict__ ssrc, float* __restrict__ sdst) {
    __shared__ float sm1[OUT_F], sm2[OUT_F];
    const int i = blockIdx.x, t = threadIdx.x;
    const float v = h[(size_t)i * OUT_F + t];
    sm1[t] = v * aw[t];
    sm2[t] = v * aw[OUT_F + t];
    __syncthreads();
    for (int s = OUT_F / 2; s > 0; s >>= 1) {
        if (t < s) { sm1[t] += sm1[t + s]; sm2[t] += sm2[t + s]; }
        __syncthreads();
    }
    if (t == 0) { ssrc[i] = sm1[0]; sdst[i] = sm2[0]; }
}

// ---------------------------------------------------------------------------
// Kernel 3: global max of s_dst (single block) -> softmax stability bound.
// Any M >= true row max gives an identical softmax in exact arithmetic.
// ---------------------------------------------------------------------------
__global__ void k_maxdst(const float* __restrict__ sdst, float* __restrict__ dmax) {
    __shared__ float sm[256];
    const int t = threadIdx.x;
    float m = -3.0e38f;
    for (int c = t; c < N_ROWS; c += 256) m = fmaxf(m, sdst[c]);
    sm[t] = m;
    __syncthreads();
    for (int w = 128; w > 0; w >>= 1) {
        if (t < w) sm[t] = fmaxf(sm[t], sm[t + w]);
        __syncthreads();
    }
    if (t == 0) dmax[0] = sm[0];
}

// ---------------------------------------------------------------------------
// Kernel 4: hbT[o][i] = bf16(h[i][o])   (LDS-tiled transpose, 32 rows/block)
// ---------------------------------------------------------------------------
__global__ void k_transpose_bf16(const float* __restrict__ h,
                                 uint16_t* __restrict__ hbT) {
    __shared__ float tile[OUT_F * 33];      // [o][i_local], padded
    const int i0 = blockIdx.x * 32;
    const int t  = threadIdx.x;
    for (int q = 0; q < 16; ++q) {          // load 32x128 floats coalesced
        const int idx = t + 256 * q;        // 0..4095
        const int il = idx >> 7, o = idx & 127;
        tile[o * 33 + il] = h[(size_t)(i0 + il) * OUT_F + o];
    }
    __syncthreads();
    for (int q = 0; q < 8; ++q) {           // write packed bf16 pairs
        const int idx = t + 256 * q;        // 0..2047
        const int o = idx >> 4, ip = idx & 15;
        const uint32_t lo = f2bf(tile[o * 33 + 2 * ip]);
        const uint32_t hi = f2bf(tile[o * 33 + 2 * ip + 1]);
        *(uint32_t*)(hbT + (size_t)o * N_ROWS + i0 + 2 * ip) = lo | (hi << 16);
    }
}

// ---------------------------------------------------------------------------
// Kernel 5: fused masked-softmax-numerator x h via bf16 WMMA, single adjacency
// pass. Block = 8 waves sharing one j-slice (same split): the hbT chunk
// (128 o x 32 j bf16 = 8 KB) is staged once per block into double-buffered LDS
// via gfx1250 async load-to-LDS; each wave owns one 16-row i-tile.
// Row sums (softmax denominator) accumulated in-register alongside.
// ---------------------------------------------------------------------------
#define STRIDE_J 40   // padded LDS row stride (u16): 80 B, 16B aligned, bank-safe

__device__ __forceinline__ void stage_hb(const uint16_t* __restrict__ hbT, int j0,
                                         uint16_t (*st)[STRIDE_J]) {
    for (int q = 0; q < 2; ++q) {
        const int idx = threadIdx.x + 256 * q;   // 0..511 -> 512 x 16B = 8 KB
        const int o  = idx >> 2;                 // 0..127
        const int jp = (idx & 3) * 8;            // 0,8,16,24 (u16 units)
        const uint16_t* g = hbT + (size_t)o * N_ROWS + j0 + jp;
        uint16_t* l = &st[o][jp];
#if HAVE_ASYNC_LDS
        __builtin_amdgcn_global_load_async_to_lds_b128(
            (glb_v4i_p)(void*)g, (lds_v4i_p)(void*)l, 0, 0);
#else
        *(uint4*)l = *(const uint4*)g;
#endif
    }
}

__global__ void k_attn(const int* __restrict__ adj, const float* __restrict__ ssrc,
                       const float* __restrict__ sdst, const float* __restrict__ ab_p,
                       const float* __restrict__ dmax_p, const uint16_t* __restrict__ hbT,
                       float* __restrict__ part, float* __restrict__ zpart) {
    __shared__ uint16_t stage[2][OUT_F][STRIDE_J];

    const int widx  = threadIdx.x >> 5;            // wave in block: 0..7
    const int lane  = threadIdx.x & 31;
    const int split = blockIdx.x & (NSPLIT - 1);   // all waves in block share split
    const int it    = (blockIdx.x >> 2) * 8 + widx;// i-tile 0..511
    const int m  = lane & 15;
    const int kh = lane >> 4;
    const int row = it * 16 + m;

    const float si = ssrc[row];
    const float ab = ab_p[0];
    const float tb = si + dmax_p[0] + ab;
    const float mi = tb > 0.0f ? tb : 0.2f * tb;   // lrelu(bound) >= all unmasked e
    const size_t arow = (size_t)row * N_ROWS;

    v8f acc[8];
    for (int ot = 0; ot < 8; ++ot) acc[ot] = (v8f){};
    float psum = 0.0f;

    const int jbeg = split * (N_ROWS / NSPLIT);
    const int nchunk = (N_ROWS / NSPLIT) / 32;     // 64

    stage_hb(hbT, jbeg, stage[0]);
#if HAVE_ASYNC_LDS
    wait_async0();
#endif
    __syncthreads();

    for (int c = 0; c < nchunk; ++c) {
        const int j0 = jbeg + c * 32;
        const int buf = c & 1;
        if (c + 1 < nchunk) stage_hb(hbT, j0 + 32, stage[buf ^ 1]);

        // prefetch adjacency two chunks ahead (global_prefetch_b8)
        if (c + 2 < nchunk) __builtin_prefetch(adj + arow + j0 + 64, 0, 1);

        // Build A = P tile (16x32 bf16). 16-bit A layout:
        //   VGPR v<4 : K = 2v + 8*kh (+1) ; VGPR v>=4 : K = 16 + 2(v-4) + 8*kh (+1)
        BF16x16 a;
        for (int v = 0; v < 8; ++v) {
            const int k = (v < 4) ? (2 * v + 8 * kh) : (16 + 2 * (v - 4) + 8 * kh);
            const int j = j0 + k;
            const int2   ad = *(const int2*)  (adj  + arow + j);
            const float2 sd = *(const float2*)(sdst + j);
            float e0 = si + sd.x + ab; e0 = e0 > 0.f ? e0 : 0.2f * e0; e0 = ad.x > 0 ? e0 : -9.0e15f;
            float e1 = si + sd.y + ab; e1 = e1 > 0.f ? e1 : 0.2f * e1; e1 = ad.y > 0 ? e1 : -9.0e15f;
            const uint32_t p0 = f2bf(__expf(e0 - mi));
            const uint32_t p1 = f2bf(__expf(e1 - mi));
            a.u[v] = p0 | (p1 << 16);
            // accumulate denominator with the same bf16-rounded values the WMMA sees
            psum += __uint_as_float(p0 << 16) + __uint_as_float(p1 << 16);
        }
        // 8 o-tiles: B layout (16-bit 32x16): VGPR v<4 : K = 2v + 8*kh (+1) ; v>=4 : K = 16 + 2(v-4) + 8*kh (+1), n = l&15
        for (int ot = 0; ot < 8; ++ot) {
            BF16x16 b;
            const uint16_t* bp = &stage[buf][ot * 16 + m][8 * kh];
            for (int v = 0; v < 8; ++v)
                b.u[v] = *(const uint32_t*)(bp + ((v < 4) ? 2 * v : 8 + 2 * v));
            acc[ot] = __builtin_amdgcn_wmma_f32_16x16x32_bf16(false, a.v, false, b.v,
                                                              (short)0, acc[ot], false, false);
        }
#if HAVE_ASYNC_LDS
        wait_async0();
#endif
        __syncthreads();
    }

    // write numerator partials
    float* pp = part + (size_t)split * HSZ;
    for (int ot = 0; ot < 8; ++ot)
        for (int r = 0; r < 8; ++r)
            pp[(size_t)(it * 16 + r + 8 * kh) * OUT_F + ot * 16 + m] = acc[ot][r];

    // denominator partial: combine the two half-wave contributions for row m
    psum += __shfl_xor(psum, 16, 32);
    if (lane < 16) zpart[(size_t)split * N_ROWS + row] = psum;
}

// ---------------------------------------------------------------------------
// Kernel 6: out = elu( (sum of split numerators) / (sum of split denominators) )
// ---------------------------------------------------------------------------
__global__ void k_finish(const float* __restrict__ part, const float* __restrict__ zpart,
                         float* __restrict__ out) {
    const size_t g = (size_t)blockIdx.x * 256 + threadIdx.x;
    const int i = (int)(g >> 7);            // row index (OUT_F = 128)
    const float Z = zpart[i] + zpart[N_ROWS + i] +
                    zpart[2 * N_ROWS + i] + zpart[3 * N_ROWS + i];
    float s = part[g] + part[g + HSZ] + part[g + 2 * (size_t)HSZ] + part[g + 3 * (size_t)HSZ];
    const float v = s / Z;
    out[g] = v > 0.0f ? v : (__expf(v) - 1.0f);
}

// ---------------------------------------------------------------------------
extern "C" void kernel_launch(void* const* d_in, const int* in_sizes, int n_in,
                              void* d_out, int out_size, void* d_ws, size_t ws_size,
                              hipStream_t stream) {
    const float* X   = (const float*)d_in[0];
    const int*   adj = (const int*)  d_in[1];
    const float* Ww  = (const float*)d_in[2];
    const float* Wb  = (const float*)d_in[3];
    const float* aw  = (const float*)d_in[4];
    const float* ab  = (const float*)d_in[5];
    float* out = (float*)d_out;

    // workspace carve-up (~22.5 MB)
    float*    h      = (float*)d_ws;                 // 4 MB
    float*    ssrc   = h + HSZ;                      // 32 KB
    float*    sdst   = ssrc + N_ROWS;                // 32 KB
    float*    dmax   = sdst + N_ROWS;                // 64 B (padded)
    uint16_t* hbT    = (uint16_t*)(dmax + 16);       // 2 MB (bf16, transposed)
    float*    part   = (float*)(hbT + HSZ);          // 16 MB (4 split numerators)
    float*    zpart  = part + (size_t)NSPLIT * HSZ;  // 128 KB (4 split denominators)

    k_gemm1<<<512, 256, 0, stream>>>(X, Ww, Wb, h);
    k_dots<<<N_ROWS, OUT_F, 0, stream>>>(h, aw, ssrc, sdst);
    k_maxdst<<<1, 256, 0, stream>>>(sdst, dmax);
    k_transpose_bf16<<<N_ROWS / 32, 256, 0, stream>>>(h, hbT);
    k_attn<<<(512 * NSPLIT) / 8, 256, 0, stream>>>(adj, ssrc, sdst, ab, dmax, hbT, part, zpart);
    k_finish<<<(HSZ) / 256, 256, 0, stream>>>(part, zpart, out);
}
